// GlobalAttention_5368709120205
// MI455X (gfx1250) — matrix operand order repaired and checked by reading, not compiled, not run on hardware
//
#include <hip/hip_runtime.h>
#include <hip/hip_bf16.h>

typedef __attribute__((ext_vector_type(16))) _Float16 v16h;
typedef __attribute__((ext_vector_type(8)))  _Float16 v8h;
typedef __attribute__((ext_vector_type(8)))  float    v8f;
typedef __attribute__((ext_vector_type(4)))  float    v4f;
typedef __attribute__((ext_vector_type(4)))  int      v4i;

#define WMMA_F16(a, b, c) \
  __builtin_amdgcn_wmma_f32_16x16x32_f16(false, (a), false, (b), (short)0, (c), false, false)

// gfx1250 async global->LDS path (ASYNCcnt), with portable fallback.
#if defined(__AMDGCN__) && __has_builtin(__builtin_amdgcn_global_load_async_to_lds_b128) && \
    __has_builtin(__builtin_amdgcn_s_wait_asynccnt)
#define USE_ASYNC_LDS 1
typedef __attribute__((address_space(1))) v4i gv4i;  // global (AS1) int4
typedef __attribute__((address_space(3))) v4i lv4i;  // LDS    (AS3) int4
#else
#define USE_ASYNC_LDS 0
#endif

// ---------------------------------------------------------------------------
// WMMA fragment loaders (ISA 7.12.2 layouts, wave32)
//   A 16x32 f16: lane (g=lane>>4, m=lane&15) holds K = {g*8..g*8+7} and
//                {16+g*8..} of row m  -> two contiguous 16B loads.
//   B 32x16 f16: lane (g, n=lane&15) holds K = {g*8..g*8+7} and
//                {16+g*8..} of column n -> two 16B loads from B^T[n][k].
//   C/D 16x16 f32: element r of lane (g,n) = C[8*g + r][n].
// ---------------------------------------------------------------------------
__device__ __forceinline__ v16h load_afrag(const _Float16* rowptr, int k0, int g) {
  v8h lo = *(const v8h*)(rowptr + k0 + g * 8);
  v8h hi = *(const v8h*)(rowptr + k0 + 16 + g * 8);
  v16h a;
#pragma unroll
  for (int j = 0; j < 8; ++j) { a[j] = lo[j]; a[j + 8] = hi[j]; }
  return a;
}

__device__ __forceinline__ v16h load_bfrag(const _Float16* colptr, int k0, int g) {
  v8h lo = *(const v8h*)(colptr + k0 + g * 8);
  v8h hi = *(const v8h*)(colptr + k0 + 16 + g * 8);
  v16h b;
#pragma unroll
  for (int j = 0; j < 8; ++j) { b[j] = lo[j]; b[j + 8] = hi[j]; }
  return b;
}

// ---------------------------------------------------------------------------
// Elementwise / transpose / im2col helpers
// ---------------------------------------------------------------------------
__global__ void cvt_f32_f16_x4(const float* __restrict__ src,
                               _Float16* __restrict__ dst, int n4) {
  int i = blockIdx.x * blockDim.x + threadIdx.x;
  if (i >= n4) return;
  v4f v = *(const v4f*)(src + (size_t)i * 4);
  _Float16* d = dst + (size_t)i * 4;
  d[0] = (_Float16)v.x; d[1] = (_Float16)v.y;
  d[2] = (_Float16)v.z; d[3] = (_Float16)v.w;
}

// dst[n*K + k] = (f16) src[k*N + n]
__global__ void transpose_cvt(const float* __restrict__ src,
                              _Float16* __restrict__ dst, int K, int N) {
  int i = blockIdx.x * blockDim.x + threadIdx.x;
  if (i >= K * N) return;
  int n = i / K, k = i - n * K;
  dst[i] = (_Float16)src[(size_t)k * N + n];
}

// akv[(b*64 + oy*8+ox)*12544 + (ky*7+kx)*256 + c] = xh[b, oy*7+ky, ox*7+kx, c]
__global__ void im2col_kv(const _Float16* __restrict__ xh,
                          _Float16* __restrict__ akv) {
  size_t i = (size_t)blockIdx.x * blockDim.x + threadIdx.x;  // 1024*12544
  int row = (int)(i / 12544);
  int kk  = (int)(i - (size_t)row * 12544);
  int b = row >> 6, p = row & 63, oy = p >> 3, ox = p & 7;
  int kyx = kk >> 8, c = kk & 255;
  int ky = kyx / 7, kx = kyx - ky * 7;
  int yy = oy * 7 + ky, xx = ox * 7 + kx;
  akv[i] = xh[(((size_t)(b * 56 + yy)) * 56 + xx) * 256 + c];
}

// vt[((b*8+h)*64 + d)*64 + key] = kv[(b*64+key)*1024 + 512 + h*64 + d]
__global__ void transpose_v(const _Float16* __restrict__ kv,
                            _Float16* __restrict__ vt) {
  int i = blockIdx.x * blockDim.x + threadIdx.x;  // 524288
  if (i >= 16 * 8 * 64 * 64) return;
  int k = i & 63, d = (i >> 6) & 63, h = (i >> 12) & 7, b = i >> 15;
  vt[i] = kv[((size_t)(b * 64 + k)) * 1024 + 512 + h * 64 + d];
}

// ---------------------------------------------------------------------------
// LDS-tiled WMMA GEMM: C[M,N] = A[M,K] * B[K,N], B supplied transposed
// (Bt[N][K]). Block tile 128x128, K-slice 32, 8 waves in a 2x4 grid,
// each wave owns a 64x32 subtile (4x2 of 16x16 -> 8 WMMAs per K-slice).
// Double-buffered async global->LDS staging (ASYNCcnt) when available.
// ---------------------------------------------------------------------------
// Stage one 128x32 f16 slab (16B chunk per thread-iteration, 2 iterations).
__device__ __forceinline__ void stage_slab(const _Float16* gbase, size_t gstride,
                                           _Float16* lds, int kc, int tid) {
#pragma unroll
  for (int it = 0; it < 2; ++it) {
    int ch  = tid + it * 256;        // 0..511 chunks of 8 halves
    int row = ch >> 2;
    int c8  = (ch & 3) * 8;
    const _Float16* gp = gbase + (size_t)row * gstride + kc + c8;
    _Float16* lp = lds + row * 32 + c8;
#if USE_ASYNC_LDS
    __builtin_amdgcn_global_load_async_to_lds_b128((gv4i*)gp, (lv4i*)lp, 0, 0);
#else
    *(v8h*)lp = *(const v8h*)gp;
#endif
  }
}

template <bool F32OUT>
__global__ __launch_bounds__(256) void gemm_wmma_f16(
    const _Float16* __restrict__ A, const _Float16* __restrict__ Bt,
    _Float16* __restrict__ Ch, float* __restrict__ Cf,
    const float* __restrict__ bias, int M, int N, int K) {
  __shared__ _Float16 As[2][128 * 32];
  __shared__ _Float16 Bs[2][128 * 32];

  const int tid  = threadIdx.x;
  const int wid  = tid >> 5;
  const int wr   = wid & 1;        // 2 wave-rows  (64 rows each)
  const int wc   = wid >> 1;       // 4 wave-cols  (32 cols each)
  const int lane = tid & 31;
  const int g = lane >> 4, ln = lane & 15;

  const int ntb   = N >> 7;
  const int mtile = blockIdx.x / ntb;
  const int ntile = blockIdx.x - mtile * ntb;
  const int mrow0 = mtile * 128;
  const int ncol0 = ntile * 128;

  const _Float16* Ab = A  + (size_t)mrow0 * K;
  const _Float16* Bb = Bt + (size_t)ncol0 * K;

  v8f acc[4][2] = {};
  const int nk = K >> 5;

#if USE_ASYNC_LDS
  stage_slab(Ab, K, As[0], 0, tid);
  stage_slab(Bb, K, Bs[0], 0, tid);
  for (int i = 0; i < nk; ++i) {
    const int cur = i & 1;
    __builtin_amdgcn_s_wait_asynccnt(0);   // stage i landed in LDS
    __syncthreads();                       // visible to all waves
    if (i + 1 < nk) {                      // overlap stage i+1 with compute i
      stage_slab(Ab, K, As[cur ^ 1], (i + 1) * 32, tid);
      stage_slab(Bb, K, Bs[cur ^ 1], (i + 1) * 32, tid);
    }
    const _Float16* as = As[cur];
    const _Float16* bs = Bs[cur];
    v16h af[4], bf[2];
#pragma unroll
    for (int mf = 0; mf < 4; ++mf)
      af[mf] = load_afrag(as + (wr * 64 + mf * 16 + ln) * 32, 0, g);
#pragma unroll
    for (int nf = 0; nf < 2; ++nf)
      bf[nf] = load_bfrag(bs + (wc * 32 + nf * 16 + ln) * 32, 0, g);
#pragma unroll
    for (int mf = 0; mf < 4; ++mf)
#pragma unroll
      for (int nf = 0; nf < 2; ++nf)
        acc[mf][nf] = WMMA_F16(af[mf], bf[nf], acc[mf][nf]);
    __syncthreads();                       // finish reads before buf reuse
  }
#else
  for (int i = 0; i < nk; ++i) {
    const int cur = i & 1;
    stage_slab(Ab, K, As[cur], i * 32, tid);
    stage_slab(Bb, K, Bs[cur], i * 32, tid);
    __syncthreads();
    const _Float16* as = As[cur];
    const _Float16* bs = Bs[cur];
    v16h af[4], bf[2];
#pragma unroll
    for (int mf = 0; mf < 4; ++mf)
      af[mf] = load_afrag(as + (wr * 64 + mf * 16 + ln) * 32, 0, g);
#pragma unroll
    for (int nf = 0; nf < 2; ++nf)
      bf[nf] = load_bfrag(bs + (wc * 32 + nf * 16 + ln) * 32, 0, g);
#pragma unroll
    for (int mf = 0; mf < 4; ++mf)
#pragma unroll
      for (int nf = 0; nf < 2; ++nf)
        acc[mf][nf] = WMMA_F16(af[mf], bf[nf], acc[mf][nf]);
    __syncthreads();
  }
#endif

#pragma unroll
  for (int mf = 0; mf < 4; ++mf)
#pragma unroll
    for (int nf = 0; nf < 2; ++nf) {
      int col = ncol0 + wc * 32 + nf * 16 + ln;
#pragma unroll
      for (int r = 0; r < 8; ++r) {
        size_t idx = (size_t)(mrow0 + wr * 64 + mf * 16 + 8 * g + r) * N + col;
        if (F32OUT) Cf[idx] = acc[mf][nf][r] + bias[col];
        else        Ch[idx] = (_Float16)acc[mf][nf][r];
      }
    }
}

// ---------------------------------------------------------------------------
// Fused attention: per wave = one (batch, head, 16-query tile).
// ---------------------------------------------------------------------------
__global__ __launch_bounds__(256) void attn_wmma(
    const _Float16* __restrict__ q,    // [b*3136, 512]
    const _Float16* __restrict__ kv,   // [b*64, 1024]  (cols 0..511 = K part)
    const _Float16* __restrict__ vt,   // [((b*8+h)*64 + d), 64 keys]
    _Float16* __restrict__ out) {      // [b*3136, 512]
  __shared__ _Float16 lds[8][16][64];
  int wid  = threadIdx.x >> 5;
  int wave = blockIdx.x * 8 + wid;     // 16*8*196 = 25088 waves exactly
  int lane = threadIdx.x & 31;
  int g = lane >> 4, ln = lane & 15;
  int b   = wave / (8 * 196);
  int rem = wave - b * (8 * 196);
  int h  = rem / 196;
  int qt = rem - h * 196;

  const _Float16* qrow = q + ((size_t)(b * 3136 + qt * 16 + ln)) * 512 + h * 64;

  // scores = Q * K^T  (M=16 queries, N=64 keys, K=64 dh)
  v8f s[4] = {};
#pragma unroll
  for (int k0 = 0; k0 < 64; k0 += 32) {
    v16h a = load_afrag(qrow, k0, g);
#pragma unroll
    for (int t = 0; t < 4; ++t) {
      const _Float16* kcol = kv + ((size_t)(b * 64 + t * 16 + ln)) * 1024 + h * 64;
      v16h bf = load_bfrag(kcol, k0, g);
      s[t] = WMMA_F16(a, bf, s[t]);
    }
  }

  // softmax over 64 keys per row (row m = 8*g + r, cols across 16 lanes)
  const float scale = 0.125f;  // 64^-0.5
#pragma unroll
  for (int r = 0; r < 8; ++r) {
    float m = -3.0e38f;
#pragma unroll
    for (int t = 0; t < 4; ++t) { s[t][r] *= scale; m = fmaxf(m, s[t][r]); }
#pragma unroll
    for (int mask = 1; mask <= 8; mask <<= 1)
      m = fmaxf(m, __shfl_xor(m, mask, 32));
    float sum = 0.f;
#pragma unroll
    for (int t = 0; t < 4; ++t) { s[t][r] = __expf(s[t][r] - m); sum += s[t][r]; }
#pragma unroll
    for (int mask = 1; mask <= 8; mask <<= 1)
      sum += __shfl_xor(sum, mask, 32);
    float inv = 1.0f / sum;
#pragma unroll
    for (int t = 0; t < 4; ++t)
      lds[wid][8 * g + r][t * 16 + ln] = (_Float16)(s[t][r] * inv);
  }
  __syncthreads();

  // out = P * V  (M=16 queries, N=64 dh, K=64 keys)
  v8f o[4] = {};
  const _Float16* prow = &lds[wid][ln][0];
#pragma unroll
  for (int k0 = 0; k0 < 64; k0 += 32) {
    v16h a = load_afrag(prow, k0, g);
#pragma unroll
    for (int t = 0; t < 4; ++t) {
      const _Float16* vcol = vt + ((size_t)((b * 8 + h) * 64 + t * 16 + ln)) * 64;
      v16h bf = load_bfrag(vcol, k0, g);
      o[t] = WMMA_F16(a, bf, o[t]);
    }
  }
#pragma unroll
  for (int t = 0; t < 4; ++t)
#pragma unroll
    for (int r = 0; r < 8; ++r)
      out[((size_t)(b * 3136 + qt * 16 + 8 * g + r)) * 512 + h * 64 + t * 16 + ln] =
          (_Float16)o[t][r];
}

// ---------------------------------------------------------------------------
extern "C" void kernel_launch(void* const* d_in, const int* in_sizes, int n_in,
                              void* d_out, int out_size, void* d_ws, size_t ws_size,
                              hipStream_t stream) {
  const float* x   = (const float*)d_in[0];   // [16,56,56,256]
  const float* Wq  = (const float*)d_in[1];   // [1,1,256,512]
  const float* Wkv = (const float*)d_in[2];   // [7,7,256,1024]
  const float* Wo  = (const float*)d_in[3];   // [1,1,512,256]
  const float* bo  = (const float*)d_in[4];   // [256]
  float* outp = (float*)d_out;                // [16,56,56,256]

  const size_t NX   = 16UL * 56 * 56 * 256;   // 12,845,056
  const size_t NPIX = 16UL * 56 * 56;         // 50,176 rows
  const size_t KKV  = 7UL * 7 * 256;          // 12,544

  char* ws = (char*)d_ws;
  size_t off = 0;
  auto alloc = [&](size_t bytes) {
    char* p = ws + off;
    off = (off + bytes + 255) & ~(size_t)255;
    return p;
  };
  _Float16* xh     = (_Float16*)alloc(NX * 2);            // x in f16
  _Float16* wqt    = (_Float16*)alloc(512UL * 256 * 2);   // Wq^T  [512][256]
  _Float16* wkvt   = (_Float16*)alloc(1024UL * KKV * 2);  // Wkv^T [1024][12544]
  _Float16* wot    = (_Float16*)alloc(256UL * 512 * 2);   // Wo^T  [256][512]
  _Float16* akv    = (_Float16*)alloc(1024UL * KKV * 2);  // im2col A [1024][12544]
  _Float16* qbuf   = (_Float16*)alloc(NPIX * 512 * 2);    // Q [50176][512]
  _Float16* kvbuf  = (_Float16*)alloc(1024UL * 1024 * 2); // KV [1024][1024]
  _Float16* vtbuf  = (_Float16*)alloc(16UL * 8 * 64 * 64 * 2);
  _Float16* aout   = (_Float16*)alloc(NPIX * 512 * 2);    // attn out [50176][512]
  (void)ws_size; (void)in_sizes; (void)n_in; (void)out_size;

  const int T = 256;
  cvt_f32_f16_x4<<<(int)(NX / 4 + T - 1) / T, T, 0, stream>>>(x, xh, (int)(NX / 4));
  transpose_cvt<<<(256 * 512 + T - 1) / T, T, 0, stream>>>(Wq, wqt, 256, 512);
  transpose_cvt<<<(int)((KKV * 1024 + T - 1) / T), T, 0, stream>>>(Wkv, wkvt, (int)KKV, 1024);
  transpose_cvt<<<(512 * 256 + T - 1) / T, T, 0, stream>>>(Wo, wot, 512, 256);
  im2col_kv<<<(int)((1024 * KKV + T - 1) / T), T, 0, stream>>>(xh, akv);

  // Q projection: [50176x256] x [256x512] -> qbuf (f16)
  gemm_wmma_f16<false><<<(int)(NPIX / 128) * (512 / 128), T, 0, stream>>>(
      xh, wqt, qbuf, nullptr, nullptr, (int)NPIX, 512, 256);
  // KV conv as GEMM: [1024x12544] x [12544x1024] -> kvbuf (f16)
  gemm_wmma_f16<false><<<(1024 / 128) * (1024 / 128), T, 0, stream>>>(
      akv, wkvt, kvbuf, nullptr, nullptr, 1024, 1024, (int)KKV);
  // V transpose for PV matmul B fragments
  transpose_v<<<(16 * 8 * 64 * 64 + T - 1) / T, T, 0, stream>>>(kvbuf, vtbuf);
  // fused attention
  attn_wmma<<<(16 * 8 * 196) / 8, T, 0, stream>>>(qbuf, kvbuf, vtbuf, aout);
  // output projection + bias -> f32 d_out
  gemm_wmma_f16<true><<<(int)(NPIX / 128) * (256 / 128), T, 0, stream>>>(
      aout, wot, nullptr, outp, bo, (int)NPIX, 256, 512);
}
